// S4DToy_10926396801046
// MI455X (gfx1250) — compile-verified, matrix operand order checked by reading, not run on hardware
//
#include <hip/hip_runtime.h>
#include <hip/hip_bf16.h>
#include <stdint.h>

// ---------------- problem constants ----------------
constexpr int Dm   = 1024;   // model dim
constexpr int Lseq = 1024;   // sequence length
constexpr int Bb   = 8;      // batch
constexpr int Nst  = 32;     // SSM modes (conjugate pairs)
constexpr int Mtot = Bb * Lseq;  // 8192 token positions
constexpr int Ktot = Dm;         // GEMM reduction dim
constexpr int LDST = 40;         // LDS row stride (bf16 elems): 20 dwords -> conflict-free

// ---------------- workspace layout (bytes) ----------------
constexpr size_t OFF_XBF  = 0;                                       // x as bf16        16 MB
constexpr size_t OFF_INW  = OFF_XBF  + (size_t)Mtot * Dm * 2;        // in_W bf16         2 MB
constexpr size_t OFF_GLUW = OFF_INW  + (size_t)Dm * Dm * 2;          // glu_W bf16        4 MB
constexpr size_t OFF_OUTW = OFF_GLUW + (size_t)2 * Dm * Dm * 2;      // out_W bf16        2 MB
constexpr size_t OFF_U    = OFF_OUTW + (size_t)Dm * Dm * 2;          // u = in-proj f32  32 MB
constexpr size_t OFF_Y    = OFF_U    + (size_t)Mtot * Dm * 4;        // post-GELU bf16   16 MB
constexpr size_t OFF_YG   = OFF_Y    + (size_t)Mtot * Dm * 2;        // post-GLU bf16    16 MB
constexpr size_t OFF_PAR  = OFF_YG   + (size_t)Mtot * Dm * 2;        // lambda/Ceff f4  512 KB

// ---------------- types ----------------
typedef __attribute__((ext_vector_type(16))) __bf16 v16bf;
typedef __attribute__((ext_vector_type(8)))  float  v8f;
typedef __attribute__((ext_vector_type(4)))  int    v4i;

union Frag {
    v16bf v;
    uint4 q[2];
};

// ---------------- CDNA5 async global->LDS DMA (ASYNCcnt-tracked) ----------------
// Builtin signature (from hipcc diagnostic): (int4 AS1*, int4 AS3*, imm offset, imm cpol)
#if __has_builtin(__builtin_amdgcn_global_load_async_to_lds_b128)
#define HAVE_ASYNC_LDS 1
typedef __attribute__((address_space(1))) v4i gv4i_t;
typedef __attribute__((address_space(3))) v4i lv4i_t;
static __device__ __forceinline__ void async_copy_b128(const void* g, void* l) {
    __builtin_amdgcn_global_load_async_to_lds_b128((gv4i_t*)g, (lv4i_t*)l, 0, 0);
}
static __device__ __forceinline__ void wait_async0() {
#if __has_builtin(__builtin_amdgcn_s_wait_asynccnt)
    __builtin_amdgcn_s_wait_asynccnt(0);
#else
    asm volatile("s_wait_asynccnt 0x0" ::: "memory");
#endif
}
#else
#define HAVE_ASYNC_LDS 0
#endif

static __device__ __forceinline__ uint16_t f2bf(float f) {
    uint32_t u = __float_as_uint(f);
    uint32_t r = u + 0x7FFFu + ((u >> 16) & 1u);   // round-to-nearest-even
    return (uint16_t)(r >> 16);
}

// ---------------- fp32 -> bf16 conversion ----------------
__global__ __launch_bounds__(256) void k_f32_to_bf16(const float* __restrict__ src,
                                                     uint16_t* __restrict__ dst, int n4) {
    int i = blockIdx.x * 256 + threadIdx.x;
    if (i >= n4) return;
    float4 f = reinterpret_cast<const float4*>(src)[i];
    union { uint16_t s[4]; uint2 u; } o;
    o.s[0] = f2bf(f.x); o.s[1] = f2bf(f.y); o.s[2] = f2bf(f.z); o.s[3] = f2bf(f.w);
    reinterpret_cast<uint2*>(dst)[i] = o.u;
}

// ---------------- S4D discretization: lambda = exp(dt*A), Ceff = 2*C*(exp(dtA)-1)/A ----
__global__ __launch_bounds__(256) void k_params(const float* __restrict__ log_dt,
                                                const float* __restrict__ C,
                                                const float* __restrict__ logAr,
                                                const float* __restrict__ Aim,
                                                float4* __restrict__ par) {
    int idx = blockIdx.x * 256 + threadIdx.x;
    if (idx >= Dm * Nst) return;
    int d = idx >> 5;
    float dt = expf(log_dt[d]);
    float ar = -expf(logAr[idx]);
    float ai = Aim[idx];
    float dr = ar * dt, di = ai * dt;
    float em = expf(dr);
    float lr = em * cosf(di), li = em * sinf(di);     // lambda
    float nr = lr - 1.0f, ni = li;                    // lambda - 1
    float inv = 1.0f / (ar * ar + ai * ai);
    float qr = (nr * ar + ni * ai) * inv;             // (lambda-1)/A
    float qi = (ni * ar - nr * ai) * inv;
    float cr = C[2 * idx], ci = C[2 * idx + 1];
    par[idx] = make_float4(lr, li, 2.0f * (cr * qr - ci * qi), 2.0f * (cr * qi + ci * qr));
}

// ---------------- bf16 WMMA GEMM:  out[m,o] = act( sum_k A[m,k]*W[o,k] + bias[o] ) -----
// EPI 0: out fp32 (the in-projection "u")
// EPI 1: fused GLU: two W halves (o and Dm+o), out bf16 = (a+ba)*sigmoid(g+bg)
// EPI 2: out fp32 = acc + bias + residual x
template <int EPI>
__global__ __launch_bounds__(256) void k_gemm(const uint16_t* __restrict__ A,
                                              const uint16_t* __restrict__ W,
                                              const float* __restrict__ bias,
                                              float* __restrict__ outF,
                                              uint16_t* __restrict__ outB,
                                              const float* __restrict__ resid) {
    constexpr bool DUAL = (EPI == 1);
    constexpr int NT    = DUAL ? 1 : 2;        // 16-wide N tiles per wave (per half)
    constexpr int nK    = Ktot / 32;

    __shared__ uint16_t As[2][128 * LDST];
    __shared__ uint16_t Bs[2][128 * LDST];     // dual: rows 0..63 = a-half, 64..127 = g-half

    const int tid   = threadIdx.x;
    const int lane  = tid & 31;
    const int wid   = tid >> 5;
    const int waveM = wid >> 2;                // 0..1 -> 64-row stripes
    const int waveN = wid & 3;                 // 0..3
    const int m0    = blockIdx.x * 128;
    const int o0    = blockIdx.y * (DUAL ? 64 : 128);

    // staging map: 512 (row,seg) slots over 2 passes of 256 threads (pass1 = row+64)
    const int r0 = tid >> 2;                   // 0..63
    const int sg = tid & 3;                    // 16B segment within 64B row
    const int r1 = r0 + 64;
    const int gBr0 = o0 + r0;                                            // a-half rows
    const int gBr1 = DUAL ? (Dm + o0 + (r1 - 64)) : (o0 + r1);           // g-half rows
    const size_t gAoff0 = (size_t)(m0 + r0) * Ktot + sg * 8;
    const size_t gAoff1 = (size_t)(m0 + r1) * Ktot + sg * 8;
    const size_t gBoff0 = (size_t)gBr0 * Ktot + sg * 8;
    const size_t gBoff1 = (size_t)gBr1 * Ktot + sg * 8;
    const int ldsOff0 = r0 * LDST + sg * 8;    // uint16 units
    const int ldsOff1 = r1 * LDST + sg * 8;

    v8f acc[4][NT];
    v8f accG[4];
    {
        v8f z = {0.f, 0.f, 0.f, 0.f, 0.f, 0.f, 0.f, 0.f};
        #pragma unroll
        for (int i = 0; i < 4; ++i) {
            #pragma unroll
            for (int j = 0; j < NT; ++j) acc[i][j] = z;
            accG[i] = z;
        }
    }

    const int frow  = lane & 15;
    const int fhalf = lane >> 4;

#if HAVE_ASYNC_LDS
    // -------- async global->LDS pipeline: no staging VGPRs at all --------
#define STAGE_ASYNC(kt, buf)                                                         \
    do {                                                                             \
        async_copy_b128(A + gAoff0 + (kt) * 32, &As[(buf)][ldsOff0]);                \
        async_copy_b128(A + gAoff1 + (kt) * 32, &As[(buf)][ldsOff1]);                \
        async_copy_b128(W + gBoff0 + (kt) * 32, &Bs[(buf)][ldsOff0]);                \
        async_copy_b128(W + gBoff1 + (kt) * 32, &Bs[(buf)][ldsOff1]);                \
    } while (0)

    STAGE_ASYNC(0, 0);
    wait_async0();
    __syncthreads();
#else
    // -------- fallback: register staging with named scalars (no stack arrays) -----
    uint4 ta0, ta1, tb0, tb1;
#define GLOAD(kt)                                                                    \
    do {                                                                             \
        ta0 = *reinterpret_cast<const uint4*>(A + gAoff0 + (kt) * 32);               \
        ta1 = *reinterpret_cast<const uint4*>(A + gAoff1 + (kt) * 32);               \
        tb0 = *reinterpret_cast<const uint4*>(W + gBoff0 + (kt) * 32);               \
        tb1 = *reinterpret_cast<const uint4*>(W + gBoff1 + (kt) * 32);               \
    } while (0)
#define DSTORE(buf)                                                                  \
    do {                                                                             \
        *reinterpret_cast<uint4*>(&As[(buf)][ldsOff0]) = ta0;                        \
        *reinterpret_cast<uint4*>(&As[(buf)][ldsOff1]) = ta1;                        \
        *reinterpret_cast<uint4*>(&Bs[(buf)][ldsOff0]) = tb0;                        \
        *reinterpret_cast<uint4*>(&Bs[(buf)][ldsOff1]) = tb1;                        \
    } while (0)

    GLOAD(0);
    DSTORE(0);
    GLOAD(1);
    __syncthreads();
#endif

    for (int kt = 0; kt < nK; ++kt) {
        const int buf = kt & 1;
#if HAVE_ASYNC_LDS
        if (kt + 1 < nK) STAGE_ASYNC(kt + 1, buf ^ 1);   // DMA overlaps the WMMAs below
#else
        if (kt + 1 < nK) DSTORE(buf ^ 1);
        if (kt + 2 < nK) GLOAD(kt + 2);
        if (kt + 3 < nK) {
            __builtin_prefetch(A + gAoff0 + (kt + 3) * 32, 0, 0);
            __builtin_prefetch(W + gBoff0 + (kt + 3) * 32, 0, 0);
        }
#endif

        Frag af[4];
        #pragma unroll
        for (int i = 0; i < 4; ++i) {
            const uint16_t* p = &As[buf][(waveM * 64 + i * 16 + frow) * LDST];
            af[i].q[0] = *reinterpret_cast<const uint4*>(p + fhalf * 8);
            af[i].q[1] = *reinterpret_cast<const uint4*>(p + 16 + fhalf * 8);
        }
        if constexpr (DUAL) {
            Frag ba, bg;
            const uint16_t* pa = &Bs[buf][(waveN * 16 + frow) * LDST];
            const uint16_t* pg = &Bs[buf][(64 + waveN * 16 + frow) * LDST];
            ba.q[0] = *reinterpret_cast<const uint4*>(pa + fhalf * 8);
            ba.q[1] = *reinterpret_cast<const uint4*>(pa + 16 + fhalf * 8);
            bg.q[0] = *reinterpret_cast<const uint4*>(pg + fhalf * 8);
            bg.q[1] = *reinterpret_cast<const uint4*>(pg + 16 + fhalf * 8);
            #pragma unroll
            for (int i = 0; i < 4; ++i) {
                acc[i][0] = __builtin_amdgcn_wmma_f32_16x16x32_bf16(
                    false, af[i].v, false, ba.v, (short)0, acc[i][0], false, false);
                accG[i] = __builtin_amdgcn_wmma_f32_16x16x32_bf16(
                    false, af[i].v, false, bg.v, (short)0, accG[i], false, false);
            }
        } else {
            Frag bf2[2];
            #pragma unroll
            for (int j = 0; j < 2; ++j) {
                const uint16_t* p = &Bs[buf][(waveN * 32 + j * 16 + frow) * LDST];
                bf2[j].q[0] = *reinterpret_cast<const uint4*>(p + fhalf * 8);
                bf2[j].q[1] = *reinterpret_cast<const uint4*>(p + 16 + fhalf * 8);
            }
            #pragma unroll
            for (int i = 0; i < 4; ++i)
                #pragma unroll
                for (int j = 0; j < 2; ++j)
                    acc[i][j] = __builtin_amdgcn_wmma_f32_16x16x32_bf16(
                        false, af[i].v, false, bf2[j].v, (short)0, acc[i][j], false, false);
        }

#if HAVE_ASYNC_LDS
        if (kt + 1 < nK) wait_async0();   // next tile's DMA landed in LDS
#endif
        __syncthreads();
    }

    // ---- epilogue ----  C/D layout: VGPR e -> M = e + 8*(lane>=16), N = lane&15
    const int ncol = lane & 15;
    const int mh   = (lane >> 4) << 3;

    if constexpr (EPI == 1) {
        const int o  = o0 + waveN * 16 + ncol;
        const float ba = bias[o];
        const float bg = bias[Dm + o];
        #pragma unroll
        for (int i = 0; i < 4; ++i) {
            const int mbase = m0 + waveM * 64 + i * 16 + mh;
            #pragma unroll
            for (int e = 0; e < 8; ++e) {
                const int m = mbase + e;
                float a = acc[i][0][e] + ba;
                float g = accG[i][e] + bg;
                float y = a * (1.0f / (1.0f + __expf(-g)));
                outB[(size_t)m * Dm + o] = f2bf(y);
            }
        }
    } else {
        #pragma unroll
        for (int j = 0; j < NT; ++j) {
            const int o  = o0 + waveN * 32 + j * 16 + ncol;
            const float bv = bias[o];
            #pragma unroll
            for (int i = 0; i < 4; ++i) {
                const int mbase = m0 + waveM * 64 + i * 16 + mh;
                #pragma unroll
                for (int e = 0; e < 8; ++e) {
                    const int m = mbase + e;
                    float v = acc[i][j][e] + bv;
                    if constexpr (EPI == 2) v += resid[(size_t)m * Dm + o];
                    outF[(size_t)m * Dm + o] = v;
                }
            }
        }
    }
}

// ---------------- diagonal-SSM scan + skip + erf-GELU -------------------------------
// wave per (b,d) channel, lane per mode n; 32 timesteps per chunk.
__global__ __launch_bounds__(256) void k_scan(const float* __restrict__ u,
                                              const float4* __restrict__ par,
                                              const float* __restrict__ Dskip,
                                              uint16_t* __restrict__ Y) {
    const int lane = threadIdx.x & 31;
    const int wid  = threadIdx.x >> 5;
    const int w    = blockIdx.x * 8 + wid;
    const int b    = w >> 10;            // Dm = 1024 channels per batch
    const int d    = w & (Dm - 1);

    const float4 p = par[d * Nst + lane];
    const float lr = p.x, li = p.y, cer = p.z, cei = p.w;
    const float dskip = Dskip[d];
    const float inv_sqrt2 = 0.70710678118654752f;

    float sr = 0.0f, si = 0.0f;
    for (int l0 = 0; l0 < Lseq; l0 += 32) {
        // lane t pre-loads timestep l0+t of this channel
        float uvec = u[(size_t)(b * Lseq + l0 + lane) * Dm + d];
        float keep = 0.0f;
        #pragma unroll
        for (int t = 0; t < 32; ++t) {
            float uu = __shfl(uvec, t, 32);
            float nr = sr * lr - si * li + uu;     // s = lambda*s + u
            float ni = sr * li + si * lr;
            sr = nr; si = ni;
            float c = cer * sr - cei * si;         // Re(Ceff * s), 2x folded into Ceff
            #pragma unroll
            for (int off = 16; off > 0; off >>= 1) c += __shfl_xor(c, off, 32);
            float yv = c + uu * dskip;
            yv = 0.5f * yv * (1.0f + erff(yv * inv_sqrt2));   // exact-erf GELU
            if (lane == t) keep = yv;
        }
        Y[(size_t)(b * Lseq + l0 + lane) * Dm + d] = f2bf(keep);
    }
}

// ---------------- driver ----------------
extern "C" void kernel_launch(void* const* d_in, const int* in_sizes, int n_in,
                              void* d_out, int out_size, void* d_ws, size_t ws_size,
                              hipStream_t stream) {
    const float* x        = (const float*)d_in[0];
    const float* in_W     = (const float*)d_in[1];
    const float* in_b     = (const float*)d_in[2];
    const float* log_dt   = (const float*)d_in[3];
    const float* Cri      = (const float*)d_in[4];
    const float* log_A_re = (const float*)d_in[5];
    const float* A_im     = (const float*)d_in[6];
    const float* D_skip   = (const float*)d_in[7];
    const float* glu_W    = (const float*)d_in[8];
    const float* glu_b    = (const float*)d_in[9];
    const float* out_W    = (const float*)d_in[10];
    const float* out_b    = (const float*)d_in[11];

    char* ws = (char*)d_ws;
    uint16_t* xbf    = (uint16_t*)(ws + OFF_XBF);
    uint16_t* inWbf  = (uint16_t*)(ws + OFF_INW);
    uint16_t* gluWbf = (uint16_t*)(ws + OFF_GLUW);
    uint16_t* outWbf = (uint16_t*)(ws + OFF_OUTW);
    float*    u      = (float*)   (ws + OFF_U);
    uint16_t* Ybf    = (uint16_t*)(ws + OFF_Y);
    uint16_t* Ygbf   = (uint16_t*)(ws + OFF_YG);
    float4*   par    = (float4*)  (ws + OFF_PAR);

    // 1) fp32 -> bf16 conversions
    {
        int n4;
        n4 = Mtot * Dm / 4;  k_f32_to_bf16<<<(n4 + 255) / 256, 256, 0, stream>>>(x, xbf, n4);
        n4 = Dm * Dm / 4;    k_f32_to_bf16<<<(n4 + 255) / 256, 256, 0, stream>>>(in_W, inWbf, n4);
        n4 = 2 * Dm * Dm / 4;k_f32_to_bf16<<<(n4 + 255) / 256, 256, 0, stream>>>(glu_W, gluWbf, n4);
        n4 = Dm * Dm / 4;    k_f32_to_bf16<<<(n4 + 255) / 256, 256, 0, stream>>>(out_W, outWbf, n4);
    }

    // 2) SSM discretization params
    k_params<<<(Dm * Nst + 255) / 256, 256, 0, stream>>>(log_dt, Cri, log_A_re, A_im, par);

    // 3) in-projection: u = x @ in_W^T + in_b   (fp32 out for the scan)
    {
        dim3 g(Mtot / 128, Dm / 128);
        k_gemm<0><<<g, 256, 0, stream>>>(xbf, inWbf, in_b, u, nullptr, nullptr);
    }

    // 4) diagonal-SSM scan + D_skip + erf-GELU -> Ybf
    k_scan<<<(Bb * Dm) / 8, 256, 0, stream>>>(u, par, D_skip, Ybf);

    // 5) fused GLU GEMM: (Y @ glu_W^T + glu_b) -> a*sigmoid(g) -> Ygbf
    {
        dim3 g(Mtot / 128, Dm / 64);
        k_gemm<1><<<g, 256, 0, stream>>>(Ybf, gluWbf, glu_b, nullptr, Ygbf, nullptr);
    }

    // 6) out-projection + residual: out = Yg @ out_W^T + out_b + x
    {
        dim3 g(Mtot / 128, Dm / 128);
        k_gemm<2><<<g, 256, 0, stream>>>(Ygbf, outWbf, out_b, (float*)d_out, nullptr, x);
    }
}
